// FastRCNN_20083267076516
// MI455X (gfx1250) — matrix operand order repaired and checked by reading, not compiled, not run on hardware
//
#include <hip/hip_runtime.h>
#include <hip/hip_bf16.h>

// ---------------------------------------------------------------------------
// Fast-RCNN head for MI455X (gfx1250, wave32, WMMA).
// Pipeline: convert weights -> matcher -> roipool (emit bf16) ->
//           GEMM1 (4608->1024, bf16 WMMA, bias+relu) ->
//           GEMM2 (1024->1024, bf16 WMMA, bias+relu) ->
//           head GEMM (1024->100 fused cls+box, f32 out)
// All intermediates live in d_ws (L2-resident, ~37 MB).
// ---------------------------------------------------------------------------

typedef __bf16 bf16_t;
typedef __attribute__((ext_vector_type(16))) __bf16 v16bf;
typedef __attribute__((ext_vector_type(8)))  __bf16 v8bf;
typedef __attribute__((ext_vector_type(8)))  float  v8f;

#define NB   4
#define NR   512
#define NG   16
#define NP   528          // R+G
#define NC   512          // channels
#define HF   25
#define WF   25
#define POOLN 3
#define KDIM (NC*POOLN*POOLN)   // 4608
#define HID  1024
#define NCLS 20
#define NROI (NB*NR)            // 2048
#define NHEAD 100               // 20 cls + 80 box
#define NHEADP 112              // padded to 7 wmma tiles

#define BM 128
#define BN 128
#define BK 32
#define LDS_STRIDE 40           // bf16 elements; 80B rows break bank conflicts

static __device__ __forceinline__ v16bf cat8(v8bf lo, v8bf hi) {
    return __builtin_shufflevector(lo, hi, 0,1,2,3,4,5,6,7,8,9,10,11,12,13,14,15);
}

// ---------------------------------------------------------------------------
// 1) Weight conversion: W1,W2 -> bf16; fuse Wc(20)+Ws(80) into 112-row padded
//    bf16 matrix; fuse bc+bs into bcs[112] floats.
// ---------------------------------------------------------------------------
__global__ __launch_bounds__(256) void convert_weights_kernel(
    const float* __restrict__ W1, const float* __restrict__ W2,
    const float* __restrict__ Wc, const float* __restrict__ Ws,
    const float* __restrict__ bc, const float* __restrict__ bs,
    bf16_t* __restrict__ W1b, bf16_t* __restrict__ W2b,
    bf16_t* __restrict__ Wcsb, float* __restrict__ bcs)
{
    const long N1 = (long)HID * KDIM;          // 4,718,592
    const long N2 = (long)HID * HID;           // 1,048,576
    const long N3 = (long)NHEADP * HID;        //   114,688
    long i = (long)blockIdx.x * blockDim.x + threadIdx.x;
    if (i < N1) {
        W1b[i] = (bf16_t)W1[i];
    } else if (i < N1 + N2) {
        long j = i - N1;
        W2b[j] = (bf16_t)W2[j];
    } else if (i < N1 + N2 + N3) {
        long j = i - N1 - N2;
        int r = (int)(j >> 10), k = (int)(j & 1023);
        float v = 0.0f;
        if (r < NCLS)       v = Wc[r * HID + k];
        else if (r < NHEAD) v = Ws[(r - NCLS) * HID + k];
        Wcsb[j] = (bf16_t)v;
    } else if (i < N1 + N2 + N3 + NHEADP) {
        int r = (int)(i - N1 - N2 - N3);
        float v = 0.0f;
        if (r < NCLS)       v = bc[r];
        else if (r < NHEAD) v = bs[r - NCLS];
        bcs[r] = v;
    }
}

// ---------------------------------------------------------------------------
// 2) Matcher: torchvision Matcher with fg=bg=0.5; proposals = cat(rois, gt).
//    One thread per (image, proposal). Outputs written as float.
// ---------------------------------------------------------------------------
__global__ __launch_bounds__(256) void matcher_kernel(
    const float* __restrict__ rois, const float* __restrict__ gtb,
    const int* __restrict__ gtl,
    float* __restrict__ labels_out, float* __restrict__ matched_out)
{
    int idx = blockIdx.x * blockDim.x + threadIdx.x;
    if (idx >= NB * NP) return;
    int b = idx / NP, p = idx % NP;

    float px1, py1, px2, py2;
    if (p < NR) {
        const float* r = rois + ((size_t)b * NR + p) * 4;
        px1 = r[0]; py1 = r[1]; px2 = r[2]; py2 = r[3];
    } else {
        const float* r = gtb + ((size_t)b * NG + (p - NR)) * 4;
        px1 = r[0]; py1 = r[1]; px2 = r[2]; py2 = r[3];
    }
    float parea = (px2 - px1) * (py2 - py1);

    float best = -1.0f; int bi = 0;
    for (int g = 0; g < NG; ++g) {
        const float* gr = gtb + ((size_t)b * NG + g) * 4;
        float gx1 = gr[0], gy1 = gr[1], gx2 = gr[2], gy2 = gr[3];
        float garea = (gx2 - gx1) * (gy2 - gy1);
        float lx = fmaxf(gx1, px1), ly = fmaxf(gy1, py1);
        float rx = fminf(gx2, px2), ry = fminf(gy2, py2);
        float iw = fmaxf(rx - lx, 0.0f), ih = fmaxf(ry - ly, 0.0f);
        float inter = iw * ih;
        float iou = inter / (garea + parea - inter);
        if (iou > best) { best = iou; bi = g; }   // first-max like argmax
    }
    int match = (best < 0.5f) ? -1 : bi;          // HIGH_T == LOW_T -> no BETWEEN
    int clamped = match < 0 ? 0 : match;
    int label = gtl[b * NG + clamped];
    if (match < 0) label = 0;
    labels_out[idx]  = (float)label;
    matched_out[idx] = (float)clamped;
}

// ---------------------------------------------------------------------------
// 3) RoIPool (torchvision semantics), emits bf16 activations [2048 x 4608].
//    One block per ROI, threads stride channels.
// ---------------------------------------------------------------------------
__global__ __launch_bounds__(256) void roipool_kernel(
    const float* __restrict__ feat, const float* __restrict__ rois,
    bf16_t* __restrict__ Xp)
{
    const int roi = blockIdx.x;          // 0..2047
    const int b = roi >> 9;              // /512
    const float* r = rois + (size_t)roi * 4;
    const float scale = 1.0f / 32.0f;
    int x1 = (int)rintf(r[0] * scale);   // jnp.round = round-half-even
    int y1 = (int)rintf(r[1] * scale);
    int x2 = (int)rintf(r[2] * scale);
    int y2 = (int)rintf(r[3] * scale);
    float rw = (float)max(x2 - x1 + 1, 1);
    float rh = (float)max(y2 - y1 + 1, 1);

    int hs[POOLN], he[POOLN], wsx[POOLN], wex[POOLN];
#pragma unroll
    for (int p = 0; p < POOLN; ++p) {
        int a0 = (int)floorf((float)p * rh / POOLN) + y1;
        int a1 = (int)ceilf((float)(p + 1) * rh / POOLN) + y1;
        hs[p] = min(max(a0, 0), HF);
        he[p] = min(max(a1, 0), HF);
        int b0 = (int)floorf((float)p * rw / POOLN) + x1;
        int b1 = (int)ceilf((float)(p + 1) * rw / POOLN) + x1;
        wsx[p] = min(max(b0, 0), WF);
        wex[p] = min(max(b1, 0), WF);
    }

    const float* fb = feat + (size_t)b * NC * HF * WF;
    for (int c = threadIdx.x; c < NC; c += blockDim.x) {
        const float* fc = fb + (size_t)c * HF * WF;
#pragma unroll
        for (int ph = 0; ph < POOLN; ++ph) {
#pragma unroll
            for (int pw = 0; pw < POOLN; ++pw) {
                float m = 0.0f;
                if (he[ph] > hs[ph] && wex[pw] > wsx[pw]) {
                    m = -3.402823466e38f;
                    for (int h = hs[ph]; h < he[ph]; ++h)
                        for (int w = wsx[pw]; w < wex[pw]; ++w)
                            m = fmaxf(m, fc[h * WF + w]);
                }
                Xp[(size_t)roi * KDIM + c * 9 + ph * 3 + pw] = (bf16_t)m;
            }
        }
    }
}

// ---------------------------------------------------------------------------
// 4) Tiled bf16 WMMA GEMM: C[M,N] = relu(A[M,K] * Bw[N,K]^T + bias), bf16 out.
//    256 threads = 8 waves (2x4); 128x128x32 block tile in LDS; each wave owns
//    a 64x32 sub-tile (8 x v8f accumulators, 8 wmma per K-step).
// ---------------------------------------------------------------------------
__global__ __launch_bounds__(256) void gemm_bf16_relu_kernel(
    const bf16_t* __restrict__ A, int lda,
    const bf16_t* __restrict__ Bw, int ldb,
    const float* __restrict__ bias,
    bf16_t* __restrict__ C, int ldc,
    int K)
{
    __shared__ bf16_t As[BM * LDS_STRIDE];
    __shared__ bf16_t Bs[BN * LDS_STRIDE];

    const int tid  = threadIdx.x;
    const int lane = tid & 31;
    const int wid  = tid >> 5;          // 0..7
    const int wm   = wid >> 2;          // 0..1 -> 64 rows each
    const int wn   = wid & 3;           // 0..3 -> 32 cols each
    const int m0   = blockIdx.y * BM;
    const int n0   = blockIdx.x * BN;

    const int crow = tid >> 1;          // 0..127  (copy row)
    const int cseg = (tid & 1) * 16;    // 0 / 16  (copy segment, elems)

    const int lrow = lane & 15;
    const int ka   = (lane < 16) ? 0 : 8;    // A: K {0..7,16..23} / {8..15,24..31}
    const int kb   = (lane < 16) ? 0 : 8;    // B: K {0..7,16..23} / {8..15,24..31}

    v8f acc[4][2];
#pragma unroll
    for (int i = 0; i < 4; ++i)
#pragma unroll
        for (int j = 0; j < 2; ++j)
#pragma unroll
            for (int e = 0; e < 8; ++e) acc[i][j][e] = 0.0f;

    for (int k0 = 0; k0 < K; k0 += BK) {
        __syncthreads();
        {   // stage 128x32 A tile and 128x32 B tile (Bw row-major [N,K])
            const bf16_t* gA = A  + (size_t)(m0 + crow) * lda + k0 + cseg;
            const bf16_t* gB = Bw + (size_t)(n0 + crow) * ldb + k0 + cseg;
            *(uint4*)&As[crow * LDS_STRIDE + cseg]     = *(const uint4*)gA;
            *(uint4*)&As[crow * LDS_STRIDE + cseg + 8] = *(const uint4*)(gA + 8);
            *(uint4*)&Bs[crow * LDS_STRIDE + cseg]     = *(const uint4*)gB;
            *(uint4*)&Bs[crow * LDS_STRIDE + cseg + 8] = *(const uint4*)(gB + 8);
            if (k0 + BK < K) {                        // gfx1250 global_prefetch_b8
                __builtin_prefetch(gA + BK, 0, 1);
                __builtin_prefetch(gB + BK, 0, 1);
            }
        }
        __syncthreads();

        v16bf afrag[4], bfrag[2];
#pragma unroll
        for (int i = 0; i < 4; ++i) {
            int row = wm * 64 + i * 16 + lrow;
            v8bf lo = *(const v8bf*)&As[row * LDS_STRIDE + ka];
            v8bf hi = *(const v8bf*)&As[row * LDS_STRIDE + ka + 16];
            afrag[i] = cat8(lo, hi);
        }
#pragma unroll
        for (int j = 0; j < 2; ++j) {
            int col = wn * 32 + j * 16 + lrow;
            v8bf lo = *(const v8bf*)&Bs[col * LDS_STRIDE + kb];
            v8bf hi = *(const v8bf*)&Bs[col * LDS_STRIDE + kb + 16];
            bfrag[j] = cat8(lo, hi);
        }
#pragma unroll
        for (int i = 0; i < 4; ++i)
#pragma unroll
            for (int j = 0; j < 2; ++j)
                acc[i][j] = __builtin_amdgcn_wmma_f32_16x16x32_bf16(
                    false, afrag[i], false, bfrag[j],
                    (short)0, acc[i][j], false, false);
    }

    // Epilogue: C/D layout -> VGPR r: M = r + (lane<16 ? 0 : 8), N = lane%16.
#pragma unroll
    for (int i = 0; i < 4; ++i) {
#pragma unroll
        for (int j = 0; j < 2; ++j) {
            int col = n0 + wn * 32 + j * 16 + lrow;
            float bv = bias[col];
            int rbase = m0 + wm * 64 + i * 16 + ((lane < 16) ? 0 : 8);
#pragma unroll
            for (int rr = 0; rr < 8; ++rr) {
                float v = acc[i][j][rr] + bv;
                v = fmaxf(v, 0.0f);
                C[(size_t)(rbase + rr) * ldc + col] = (bf16_t)v;
            }
        }
    }
}

// ---------------------------------------------------------------------------
// 5) Head GEMM: [2048x1024] x [112x1024]^T, fragments streamed from L2.
//    One 16x16 tile per wave; cols <20 -> cls_logits, 20..99 -> box_deltas.
// ---------------------------------------------------------------------------
__global__ __launch_bounds__(128) void head_gemm_kernel(
    const bf16_t* __restrict__ H, const bf16_t* __restrict__ Wcs,
    const float* __restrict__ bcs,
    float* __restrict__ cls, float* __restrict__ box)
{
    const int lane  = threadIdx.x & 31;
    const int wid   = threadIdx.x >> 5;
    const int mtile = blockIdx.y * 4 + wid;   // 0..127
    const int ntile = blockIdx.x;             // 0..6
    const int lrow  = lane & 15;
    const int ka    = (lane < 16) ? 0 : 8;
    const int kb    = (lane < 16) ? 0 : 8;

    const bf16_t* arow = H   + (size_t)(mtile * 16 + lrow) * HID;
    const bf16_t* brow = Wcs + (size_t)(ntile * 16 + lrow) * HID;

    v8f acc;
#pragma unroll
    for (int e = 0; e < 8; ++e) acc[e] = 0.0f;

    for (int k0 = 0; k0 < HID; k0 += 32) {
        v8bf alo = *(const v8bf*)(arow + k0 + ka);
        v8bf ahi = *(const v8bf*)(arow + k0 + ka + 16);
        v8bf blo = *(const v8bf*)(brow + k0 + kb);
        v8bf bhi = *(const v8bf*)(brow + k0 + kb + 16);
        acc = __builtin_amdgcn_wmma_f32_16x16x32_bf16(
            false, cat8(alo, ahi), false, cat8(blo, bhi),
            (short)0, acc, false, false);
    }

    int col = ntile * 16 + lrow;              // < 112; rows >=100 are zero-pad
    float bv = bcs[col];
    int rbase = mtile * 16 + ((lane < 16) ? 0 : 8);
#pragma unroll
    for (int rr = 0; rr < 8; ++rr) {
        float v = acc[rr] + bv;
        if (col < NCLS)
            cls[(size_t)(rbase + rr) * NCLS + col] = v;
        else if (col < NHEAD)
            box[(size_t)(rbase + rr) * (NCLS * 4) + (col - NCLS)] = v;
    }
}

// ---------------------------------------------------------------------------
extern "C" void kernel_launch(void* const* d_in, const int* in_sizes, int n_in,
                              void* d_out, int out_size, void* d_ws, size_t ws_size,
                              hipStream_t stream) {
    const float* feat = (const float*)d_in[0];
    const float* rois = (const float*)d_in[1];
    const float* gtb  = (const float*)d_in[2];
    const int*   gtl  = (const int*)  d_in[3];
    const float* W1   = (const float*)d_in[4];
    const float* b1   = (const float*)d_in[5];
    const float* W2   = (const float*)d_in[6];
    const float* b2   = (const float*)d_in[7];
    const float* Wc   = (const float*)d_in[8];
    const float* bc   = (const float*)d_in[9];
    const float* Wsw  = (const float*)d_in[10];
    const float* bs   = (const float*)d_in[11];

    // Output layout (flat float): cls[2048*20] | box[2048*80] | labels[4*528] | matched[4*528]
    float* out = (float*)d_out;
    float* cls = out;
    float* box = out + (size_t)NROI * NCLS;                      // +40960
    float* lab = box + (size_t)NROI * NCLS * 4;                  // +163840
    float* mid = lab + (size_t)NB * NP;                          // +2112

    // Workspace layout (bf16/f32), ~37.2 MB total, L2-resident.
    char* w = (char*)d_ws;
    bf16_t* Xp   = (bf16_t*)(w);                                 // 2048*4608*2 = 18874368
    bf16_t* W1b  = (bf16_t*)(w + 18874368);                      // 1024*4608*2 =  9437184
    bf16_t* W2b  = (bf16_t*)(w + 28311552);                      // 1024*1024*2 =  2097152
    bf16_t* Wcsb = (bf16_t*)(w + 30408704);                      //  112*1024*2 =   229376
    float*  bcs  = (float*) (w + 30638080);                      //  112*4      =      448
    bf16_t* H1   = (bf16_t*)(w + 30638528);                      // 2048*1024*2 =  4194304
    bf16_t* H2   = (bf16_t*)(w + 34832832);                      // 2048*1024*2 =  4194304
    (void)ws_size; (void)in_sizes; (void)n_in; (void)out_size;

    // 1) weight conversion / fusion
    {
        long total = (long)HID * KDIM + (long)HID * HID + (long)NHEADP * HID + NHEADP;
        int blocks = (int)((total + 255) / 256);
        convert_weights_kernel<<<blocks, 256, 0, stream>>>(
            W1, W2, Wc, Wsw, bc, bs, W1b, W2b, Wcsb, bcs);
    }
    // 2) matcher
    matcher_kernel<<<(NB * NP + 255) / 256, 256, 0, stream>>>(rois, gtb, gtl, lab, mid);
    // 3) roipool -> bf16 activations
    roipool_kernel<<<NROI, 256, 0, stream>>>(feat, rois, Xp);
    // 4) fc1 + relu : [2048,4608] x [1024,4608]^T
    {
        dim3 grid(HID / BN, NROI / BM);   // (8,16)
        gemm_bf16_relu_kernel<<<grid, 256, 0, stream>>>(
            Xp, KDIM, W1b, KDIM, b1, H1, HID, KDIM);
    }
    // 5) fc2 + relu : [2048,1024] x [1024,1024]^T
    {
        dim3 grid(HID / BN, NROI / BM);
        gemm_bf16_relu_kernel<<<grid, 256, 0, stream>>>(
            H1, HID, W2b, HID, b2, H2, HID, HID);
    }
    // 6) fused cls+box head
    {
        dim3 grid(NHEADP / 16, NROI / 16 / 4);  // (7,32), 128 threads = 4 waves
        head_gemm_kernel<<<grid, 128, 0, stream>>>(H2, Wcsb, bcs, cls, box);
    }
}
